// HeteroRGCN_46445776339628
// MI455X (gfx1250) — matrix operand order repaired and checked by reading, not compiled, not run on hardware
//
#include <hip/hip_runtime.h>
#include <hip/hip_bf16.h>

#define N_NODES   71424
#define N_NEUR    279
#define NEDGE     1142784
#define FEATD     128

typedef __attribute__((ext_vector_type(16))) __bf16 v16bf;
typedef __attribute__((ext_vector_type(8)))  __bf16 v8bf;
typedef __attribute__((ext_vector_type(8)))  float  v8f;

__device__ __forceinline__ __bf16 f2bf(float x) { return (__bf16)x; }

// h0 (bf16) [N,256] = concat(embed[n % 279], feats[n])
__global__ __launch_bounds__(256) void k_build_h0(const float* __restrict__ embed,
                                                  const float* __restrict__ feats,
                                                  __bf16* __restrict__ h) {
  int gid = blockIdx.x * blockDim.x + threadIdx.x;   // over N*128
  if (gid >= N_NODES * FEATD) return;
  int n = gid >> 7, c = gid & 127;
  h[(size_t)n * 256 + c]       = f2bf(embed[(n % N_NEUR) * 128 + c]);
  h[(size_t)n * 256 + 128 + c] = f2bf(feats[gid]);
}

// Wt[m*K + k] = bf16(W[k*M + m])  (transpose + convert; weights are tiny)
__global__ __launch_bounds__(256) void k_prep_w(const float* __restrict__ W,
                                                __bf16* __restrict__ Wt, int K, int M) {
  int gid = blockIdx.x * blockDim.x + threadIdx.x;
  if (gid >= K * M) return;
  int m = gid / K, k = gid - m * K;
  Wt[gid] = f2bf(W[k * M + m]);
}

__global__ __launch_bounds__(256) void k_zero(float* __restrict__ p, int n) {
  int gid = blockIdx.x * blockDim.x + threadIdx.x;
  if (gid < n) p[gid] = 0.f;
}

__global__ __launch_bounds__(256) void k_cvt_bf(const float* __restrict__ x,
                                                __bf16* __restrict__ y, int n) {
  int gid = blockIdx.x * blockDim.x + threadIdx.x;
  if (gid < n) y[gid] = f2bf(x[gid]);
}

// C[N,128] = A[N,K](bf16) @ W (via Wt[128][K] bf16 col-major) + bias [+resid][+leaky]
// grid.x = N/16, block = 256 (8 waves); wave w -> cols [16w,16w+16)
__global__ __launch_bounds__(256) void k_gemm128(const __bf16* __restrict__ A,
                                                 const __bf16* __restrict__ Wt,
                                                 const float*  __restrict__ bias,
                                                 const float*  __restrict__ resid,  // nullable
                                                 float*        __restrict__ Cf,     // nullable
                                                 __bf16*       __restrict__ Cb,     // nullable
                                                 int K, int leaky) {
  const int tile_m = blockIdx.x << 4;
  const int wave = threadIdx.x >> 5;
  const int lane = threadIdx.x & 31;
  const int half = lane >> 4;
  const int l16  = lane & 15;
  const int col  = (wave << 4) + l16;

  v8f acc = {};
  const __bf16* arow = A  + (size_t)(tile_m + l16) * K;
  const __bf16* wcol = Wt + (size_t)col * K;
  for (int ko = 0; ko < K; ko += 32) {
    // A fragment, contiguous layout matching B: k = half*16 + e
    v8bf alo = *(const v8bf*)(arow + ko + half * 16);
    v8bf ahi = *(const v8bf*)(arow + ko + half * 16 + 8);
    v16bf a, b;
#pragma unroll
    for (int t = 0; t < 8; ++t) { a[t] = alo[t]; a[8 + t] = ahi[t]; }
    // B fragment: lane holds column `col`, k = half*16 + e  (contiguous 32B)
    b = *(const v16bf*)(wcol + ko + half * 16);
    acc = __builtin_amdgcn_wmma_f32_16x16x32_bf16(false, a, false, b,
                                                  (short)0, acc, false, false);
  }
  const float bc = bias[col];
#pragma unroll
  for (int r = 0; r < 8; ++r) {          // C/D layout: m = 8*half + r, n = l16
    int m = tile_m + (half << 3) + r;
    float v = acc[r] + bc;
    if (resid) v += resid[(size_t)m * 128 + col];
    if (leaky) v = (v > 0.f) ? v : 0.01f * v;
    if (Cf) Cf[(size_t)m * 128 + col] = v;
    if (Cb) Cb[(size_t)m * 128 + col] = f2bf(v);
  }
}

// agg[dst] += Wh[src]  (M=128; 4 floats per thread; rows L2-resident)
__global__ __launch_bounds__(256) void k_scatter128(const float* __restrict__ Wh,
                                                    const int* __restrict__ src,
                                                    const int* __restrict__ dst,
                                                    float* __restrict__ agg) {
  long long gid = (long long)blockIdx.x * blockDim.x + threadIdx.x;
  if (gid >= (long long)NEDGE * 32) return;
  int e = (int)(gid >> 5);
  int c = ((int)gid & 31) << 2;
  int s = src[e], d = dst[e];
  const float4 v = *(const float4*)(Wh + (size_t)s * 128 + c);
  float* o = agg + (size_t)d * 128 + c;
  atomicAdd(o + 0, v.x); atomicAdd(o + 1, v.y);
  atomicAdd(o + 2, v.z); atomicAdd(o + 3, v.w);
}

// last layer per-etype GEMM: Wh2[n,{0,1}] = h(bf16,[N,128]) @ W(128x2) + b
__global__ __launch_bounds__(256) void k_gemm2(const __bf16* __restrict__ A,
                                               const float* __restrict__ W,
                                               const float* __restrict__ b,
                                               float* __restrict__ Wh2) {
  int n = blockIdx.x * blockDim.x + threadIdx.x;
  if (n >= N_NODES) return;
  const __bf16* row = A + (size_t)n * 128;
  float a0 = b[0], a1 = b[1];
#pragma unroll 8
  for (int k = 0; k < 128; ++k) {
    float x = (float)row[k];
    a0 += x * W[2 * k];
    a1 += x * W[2 * k + 1];
  }
  Wh2[2 * n] = a0; Wh2[2 * n + 1] = a1;
}

__global__ __launch_bounds__(256) void k_scatter2(const float* __restrict__ Wh2,
                                                  const int* __restrict__ src,
                                                  const int* __restrict__ dst,
                                                  float* __restrict__ agg) {
  int e = blockIdx.x * blockDim.x + threadIdx.x;
  if (e >= NEDGE) return;
  int s = src[e], d = dst[e];
  atomicAdd(&agg[2 * d],     Wh2[2 * s]);
  atomicAdd(&agg[2 * d + 1], Wh2[2 * s + 1]);
}

// out = agg(N,2) @ w_out(2,2) + b_out + Wh_elec
__global__ __launch_bounds__(256) void k_final(const float* __restrict__ agg,
                                               const float* __restrict__ w,
                                               const float* __restrict__ b,
                                               const float* __restrict__ Wh2,
                                               float* __restrict__ out) {
  int n = blockIdx.x * blockDim.x + threadIdx.x;
  if (n >= N_NODES) return;
  float a0 = agg[2 * n], a1 = agg[2 * n + 1];
  out[2 * n]     = a0 * w[0] + a1 * w[2] + b[0] + Wh2[2 * n];
  out[2 * n + 1] = a0 * w[1] + a1 * w[3] + b[1] + Wh2[2 * n + 1];
}

extern "C" void kernel_launch(void* const* d_in, const int* in_sizes, int n_in,
                              void* d_out, int out_size, void* d_ws, size_t ws_size,
                              hipStream_t stream) {
  const float* feats = (const float*)d_in[0];
  const int* srcs[2] = {(const int*)d_in[1], (const int*)d_in[3]};
  const int* dsts[2] = {(const int*)d_in[2], (const int*)d_in[4]};
  const float* embed = (const float*)d_in[5];

  // Resolve per-layer param leaves; handle sorted-key vs insertion flatten order.
  const float *w_chem[4], *b_chem[4], *w_elec[4], *b_elec[4], *w_out[4], *b_out[4];
  static const int douts[4] = {128, 128, 128, 2};
  for (int L = 0; L < 4; ++L) {
    int base = 6 + 6 * L;
    if (in_sizes[base] == douts[L]) {
      // sorted keys: b_chem, b_elec, b_out, w_chem, w_elec, w_out
      b_chem[L] = (const float*)d_in[base + 0]; b_elec[L] = (const float*)d_in[base + 1];
      b_out[L]  = (const float*)d_in[base + 2]; w_chem[L] = (const float*)d_in[base + 3];
      w_elec[L] = (const float*)d_in[base + 4]; w_out[L]  = (const float*)d_in[base + 5];
    } else {
      // insertion: w_chem, b_chem, w_elec, b_elec, w_out, b_out
      w_chem[L] = (const float*)d_in[base + 0]; b_chem[L] = (const float*)d_in[base + 1];
      w_elec[L] = (const float*)d_in[base + 2]; b_elec[L] = (const float*)d_in[base + 3];
      w_out[L]  = (const float*)d_in[base + 4]; b_out[L]  = (const float*)d_in[base + 5];
    }
  }

  // Workspace layout (all 512B aligned):
  char* ws = (char*)d_ws;
  __bf16* h_bf  = (__bf16*)(ws + 0);           // N*256 bf16 = 36,569,088 B
  float*  Wh    = (float*)(ws + 36569088);     // N*128 f32  = 36,569,088 B
  float*  agg   = (float*)(ws + 73138176);     // N*128 f32  = 36,569,088 B
  __bf16* aggbf = (__bf16*)(ws + 109707264);   // N*128 bf16 = 18,284,544 B
  __bf16* Wt    = (__bf16*)(ws + 127991808);   // 256*128 bf16 = 65,536 B
  float* out = (float*)d_out;

  const int BT = 256;
  k_build_h0<<<(N_NODES * FEATD + BT - 1) / BT, BT, 0, stream>>>(embed, feats, h_bf);

  const unsigned scat_blocks = (unsigned)(((long long)NEDGE * 32 + BT - 1) / BT);
  for (int L = 0; L < 3; ++L) {
    const int K = (L == 0) ? 256 : 128;
    k_zero<<<(N_NODES * 128 + BT - 1) / BT, BT, 0, stream>>>(agg, N_NODES * 128);
    const float* wv[2] = {w_chem[L], w_elec[L]};
    const float* bv[2] = {b_chem[L], b_elec[L]};
    for (int et = 0; et < 2; ++et) {
      k_prep_w<<<(K * 128 + BT - 1) / BT, BT, 0, stream>>>(wv[et], Wt, K, 128);
      k_gemm128<<<N_NODES / 16, BT, 0, stream>>>(h_bf, Wt, bv[et],
                                                 nullptr, Wh, nullptr, K, 0);
      k_scatter128<<<scat_blocks, BT, 0, stream>>>(Wh, srcs[et], dsts[et], agg);
    }
    // h = leaky(agg @ w_out + b_out + Wh_elec); Wh still holds the elec result
    k_cvt_bf<<<(N_NODES * 128 + BT - 1) / BT, BT, 0, stream>>>(agg, aggbf, N_NODES * 128);
    k_prep_w<<<(128 * 128 + BT - 1) / BT, BT, 0, stream>>>(w_out[L], Wt, 128, 128);
    k_gemm128<<<N_NODES / 16, BT, 0, stream>>>(aggbf, Wt, b_out[L],
                                               Wh, nullptr, h_bf, 128, 1);
  }

  // Layer 3 (dout = 2): VALU path, cost is negligible
  k_zero<<<(N_NODES * 2 + BT - 1) / BT, BT, 0, stream>>>(agg, N_NODES * 2);
  k_gemm2<<<(N_NODES + BT - 1) / BT, BT, 0, stream>>>(h_bf, w_chem[3], b_chem[3], Wh);
  k_scatter2<<<(NEDGE + BT - 1) / BT, BT, 0, stream>>>(Wh, srcs[0], dsts[0], agg);
  k_gemm2<<<(N_NODES + BT - 1) / BT, BT, 0, stream>>>(h_bf, w_elec[3], b_elec[3], Wh);
  k_scatter2<<<(NEDGE + BT - 1) / BT, BT, 0, stream>>>(Wh, srcs[1], dsts[1], agg);
  k_final<<<(N_NODES + BT - 1) / BT, BT, 0, stream>>>(agg, w_out[3], b_out[3], Wh, out);
}
